// NeuralTransformer_44478681317425
// MI455X (gfx1250) — matrix operand order repaired and checked by reading, not compiled, not run on hardware
//
#include <hip/hip_runtime.h>
#include <hip/hip_bf16.h>
#include <cmath>

typedef __attribute__((ext_vector_type(16))) _Float16 v16h;
typedef __attribute__((ext_vector_type(8)))  _Float16 v8h;
typedef __attribute__((ext_vector_type(8)))  float    v8f;

// Model constants
#define BN    64
#define PP    50
#define DD    200
#define LLY   12
#define HH    10
#define HDIM  20
#define FFD   800
#define NPAT  240
#define SQ    241
#define NRDIM 482
#define NCL   1000
#define MR    (BN*SQ)     // 15424 rows (mult of 32)
#define MRE   (BN*NPAT)   // 15360 patch rows (mult of 32)
#define DP    224         // D padded to mult-32
#define PPAD  64          // P padded to mult-32
#define ATTN_SCALE 0.223606797749979f   // 20^-0.5

// ---- fragment loaders (CDNA5 16x32 f16 A / 32x16 f16 B layouts) ----
__device__ __forceinline__ v16h load_afrag(const _Float16* p) {
    // lane: row = lane&15; elems 0..15 contiguous at +hi*16
    v8h x0 = *(const v8h*)p;
    v8h x1 = *(const v8h*)(p + 8);
    v16h f;
#pragma unroll
    for (int i = 0; i < 8; ++i) { f[i] = x0[i]; f[i + 8] = x1[i]; }
    return f;
}
__device__ __forceinline__ v16h load_bfrag(const _Float16* p) {
    // lane: col = lane&15; elems 0..15 contiguous at +hi*16
    v8h x0 = *(const v8h*)p;
    v8h x1 = *(const v8h*)(p + 8);
    v16h f;
#pragma unroll
    for (int i = 0; i < 8; ++i) { f[i] = x0[i]; f[i + 8] = x1[i]; }
    return f;
}
__device__ __forceinline__ v8f wmma16(v16h a, v16h b, v8f c) {
    return __builtin_amdgcn_wmma_f32_16x16x32_f16(false, a, false, b,
                                                  (short)0, c, false, false);
}

// ---------------------------------------------------------------------------
// WMMA GEMM: out[M,N] = A[M,K]f16 @ W[N,K]f16^T (+bias)(+resid)(+gelu)
// One wave computes a 32x32 tile (2x2 WMMA tiles): 2 b128 loads per WMMA.
// K multiple of 32; M multiple of 32.
// ---------------------------------------------------------------------------
__global__ __launch_bounds__(128) void k_gemm(
    const _Float16* __restrict__ A, int lda,
    const _Float16* __restrict__ W, int ldb,
    const float* __restrict__ bias,
    const float* __restrict__ resid, int ldr,
    float* __restrict__ outF, _Float16* __restrict__ outH, int ldo,
    int M, int N, int K, int doGelu)
{
    const int lane = threadIdx.x;
    const int nT32 = (N + 31) >> 5;
    const int tile = blockIdx.x * blockDim.y + threadIdx.y;
    if (tile >= (M >> 5) * nT32) return;         // wave-uniform exit
    const int mT = tile / nT32;
    const int nT = tile - mT * nT32;
    const int r  = lane & 15;
    const int hi = lane >> 4;

    const _Float16* ap0 = A + (size_t)(mT * 32 + r) * lda + hi * 16;
    const _Float16* ap1 = ap0 + (size_t)16 * lda;
    const int n0 = nT * 32 + r;
    const int n1 = n0 + 16;
    // clamp: garbage columns are never stored
    const _Float16* bp0 = W + (size_t)((n0 < N) ? n0 : (N - 1)) * ldb + hi * 16;
    const _Float16* bp1 = W + (size_t)((n1 < N) ? n1 : (N - 1)) * ldb + hi * 16;

    v8f a00 = {}, a01 = {}, a10 = {}, a11 = {};
    for (int kb = 0; kb < K; kb += 32) {
        const v16h af0 = load_afrag(ap0 + kb);
        const v16h af1 = load_afrag(ap1 + kb);
        const v16h bf0 = load_bfrag(bp0 + kb);
        const v16h bf1 = load_bfrag(bp1 + kb);
        a00 = wmma16(af0, bf0, a00);
        a01 = wmma16(af0, bf1, a01);
        a10 = wmma16(af1, bf0, a10);
        a11 = wmma16(af1, bf1, a11);
    }

    const float* biasp  = bias;
    const float* residp = resid;
#pragma unroll
    for (int ti = 0; ti < 2; ++ti) {
#pragma unroll
        for (int tj = 0; tj < 2; ++tj) {
            const v8f acc = (ti == 0) ? (tj == 0 ? a00 : a01)
                                      : (tj == 0 ? a10 : a11);
            const int col = (tj == 0) ? n0 : n1;
            if (col < N) {
                const float bv = biasp ? biasp[col] : 0.0f;
#pragma unroll
                for (int e = 0; e < 8; ++e) {
                    const int row = mT * 32 + ti * 16 + e + 8 * hi;
                    float v = acc[e] + bv;
                    if (residp) v += residp[(size_t)row * ldr + col];
                    if (doGelu) v  = 0.5f * v * (1.0f + erff(v * 0.70710678118f));
                    if (outF) outF[(size_t)row * ldo + col] = v;
                    else      outH[(size_t)row * ldo + col] = (_Float16)v;
                }
            }
        }
    }
}

// ---------------------------------------------------------------------------
// Fused attention per (b,h): softmax(Q K^T * scale + relbias) V with WMMA.
// S padded 241->256 (16 tiles), HD padded 20->32. 4 waves/block.
// K staged as [key][hd], V^T as [hd][key] in LDS (shared by all waves);
// each wave owns 4 query tiles and a private 8KB LDS slice for P.
// LDS: 16KB K + 16KB V^T + 4*8KB P = 64KB.
// ---------------------------------------------------------------------------
__global__ __launch_bounds__(128) void k_attn(
    const float* __restrict__ qkv,   // [B*S, 600] = [q|k|v]
    const float* __restrict__ qb,    // [200] layer slice
    const float* __restrict__ vb,    // [200] layer slice
    const float* __restrict__ rpb,   // [482,10] layer slice
    float* __restrict__ o)           // [B*S, 200]
{
    __shared__ _Float16 sK[256 * 32];   // [key][hd]
    __shared__ _Float16 sVt[32 * 256];  // [hd][key]
    __shared__ _Float16 sP[4 * 4096];   // per-wave 16x256
    const int tid  = threadIdx.x;
    const int lane = tid & 31;
    const int wid  = tid >> 5;
    const int b = blockIdx.x / HH;
    const int h = blockIdx.x % HH;
    const float* qkvB = qkv + (size_t)b * SQ * 600;
    const int r  = lane & 15;
    const int hi = lane >> 4;
    _Float16* myP = sP + wid * 4096;

    // ---- cooperative staging of K and V^T (bias added, zero padded) ----
    for (int i = tid; i < 256 * 32; i += 128) {
        const int key = i >> 5, hd = i & 31;
        float v = 0.0f;
        if (hd < HDIM && key < SQ)
            v = qkvB[(size_t)key * 600 + 200 + h * HDIM + hd];
        sK[i] = (_Float16)v;
    }
    for (int i = tid; i < 32 * 256; i += 128) {
        const int hd = i >> 8, key = i & 255;
        float v = 0.0f;
        if (hd < HDIM && key < SQ)
            v = qkvB[(size_t)key * 600 + 400 + h * HDIM + hd] + vb[h * HDIM + hd];
        sVt[i] = (_Float16)v;
    }
    __syncthreads();

    for (int qt = wid; qt < 16; qt += 4) {       // 4 query tiles per wave
        // ---- Q A-fragment (16x32, scaled, q_bias added, zero-padded) ----
        v16h qa;
        const int qrowA = qt * 16 + r;
#pragma unroll
        for (int e = 0; e < 16; ++e) {
            const int hd = hi * 16 + e;
            float v = 0.0f;
            if (hd < HDIM && qrowA < SQ)
                v = (qkvB[(size_t)qrowA * 600 + h * HDIM + hd] + qb[h * HDIM + hd]) * ATTN_SCALE;
            qa[e] = (_Float16)v;
        }
        // ---- scores: 16 tiles of Q K^T, bias + mask, running row max ----
        float mrow[8];
#pragma unroll
        for (int e = 0; e < 8; ++e) mrow[e] = -3.0e38f;
        v8f sc[16];
#pragma unroll
        for (int kt = 0; kt < 16; ++kt) {
            const int key = kt * 16 + r;
            const v16h kf = load_bfrag(sK + key * 32 + hi * 16);
            v8f c = {};
            c = wmma16(qa, kf, c);
#pragma unroll
            for (int e = 0; e < 8; ++e) {
                const int q  = qt * 16 + e + 8 * hi;
                const int k2 = kt * 16 + r;
                float v;
                if (k2 < SQ && q < SQ) {
                    int idx;
                    if (q == 0 && k2 == 0) idx = NRDIM - 1;      // 481
                    else if (q == 0)       idx = NRDIM - 3;      // 479
                    else if (k2 == 0)      idx = NRDIM - 2;      // 480
                    else                   idx = q - k2 + (NPAT - 1);
                    v = c[e] + rpb[idx * HH + h];
                } else {
                    v = -3.0e38f;
                }
                c[e] = v;
                mrow[e] = fmaxf(mrow[e], v);
            }
            sc[kt] = c;
        }
        // ---- cross-lane row max within each 16-lane half ----
#pragma unroll
        for (int e = 0; e < 8; ++e) {
            float m = mrow[e];
            m = fmaxf(m, __shfl_xor(m, 1, 32));
            m = fmaxf(m, __shfl_xor(m, 2, 32));
            m = fmaxf(m, __shfl_xor(m, 4, 32));
            m = fmaxf(m, __shfl_xor(m, 8, 32));
            mrow[e] = m;
        }
        // ---- exp, row sums, unnormalized P -> private LDS slice ----
        float sum[8];
#pragma unroll
        for (int e = 0; e < 8; ++e) sum[e] = 0.0f;
#pragma unroll
        for (int kt = 0; kt < 16; ++kt) {
#pragma unroll
            for (int e = 0; e < 8; ++e) {
                const float p = __expf(sc[kt][e] - mrow[e]);
                sum[e] += p;
                const int rloc = e + 8 * hi;
                myP[rloc * 256 + kt * 16 + r] = (_Float16)p;
            }
        }
#pragma unroll
        for (int e = 0; e < 8; ++e) {
            float s = sum[e];
            s += __shfl_xor(s, 1, 32);
            s += __shfl_xor(s, 2, 32);
            s += __shfl_xor(s, 4, 32);
            s += __shfl_xor(s, 8, 32);
            sum[e] = 1.0f / s;   // row q = qt*16 + e + 8*hi, same map as O store
        }
        // ---- O = P V : K=256 in 8 chunks, 2 N-tiles (HD padded to 32) ----
#pragma unroll
        for (int nt = 0; nt < 2; ++nt) {
            v8f oc = {};
#pragma unroll
            for (int kb2 = 0; kb2 < 256; kb2 += 32) {
                const v16h pa = load_afrag(myP + r * 256 + kb2 + hi * 16);
                const int hd = nt * 16 + r;
                const v16h vf = load_bfrag(sVt + hd * 256 + kb2 + hi * 16);
                oc = wmma16(pa, vf, oc);
            }
#pragma unroll
            for (int e = 0; e < 8; ++e) {
                const int q  = qt * 16 + e + 8 * hi;
                const int hd = nt * 16 + r;
                if (q < SQ && hd < HDIM)
                    o[((size_t)b * SQ + q) * DD + h * HDIM + hd] = oc[e] * sum[e];
            }
        }
    }
}

// ---------------------------------------------------------------------------
// LayerNorm: one wave per row, f32 in -> f16 padded out (zeros in pad)
// ---------------------------------------------------------------------------
__global__ __launch_bounds__(256) void k_ln(
    const float* __restrict__ x, const float* __restrict__ s,
    const float* __restrict__ bp, _Float16* __restrict__ out,
    int ldo, int rows, int D)
{
    const int lane = threadIdx.x & 31;
    const int row  = blockIdx.x * 8 + (threadIdx.x >> 5);
    if (row >= rows) return;
    const float* xr = x + (size_t)row * D;
    float sum = 0.0f;
    for (int c = lane; c < D; c += 32) sum += xr[c];
    for (int m = 16; m >= 1; m >>= 1) sum += __shfl_xor(sum, m, 32);
    const float mean = sum / (float)D;
    float var = 0.0f;
    for (int c = lane; c < D; c += 32) { const float d = xr[c] - mean; var += d * d; }
    for (int m = 16; m >= 1; m >>= 1) var += __shfl_xor(var, m, 32);
    const float rstd = rsqrtf(var / (float)D + 1e-6f);
    _Float16* orow = out + (size_t)row * ldo;
    for (int c = lane; c < ldo; c += 32)
        orow[c] = (c < D) ? (_Float16)((xr[c] - mean) * rstd * s[c] + bp[c]) : (_Float16)0.0f;
}

// f32 [R,K] -> f16 [R,Kp] with zero pad
__global__ void k_cvt_pad(const float* __restrict__ src, _Float16* __restrict__ dst,
                          int R, int K, int Kp)
{
    const size_t i = (size_t)blockIdx.x * blockDim.x + threadIdx.x;
    if (i >= (size_t)R * Kp) return;
    const int r = (int)(i / Kp), c = (int)(i % Kp);
    dst[i] = (c < K) ? (_Float16)src[(size_t)r * K + c] : (_Float16)0.0f;
}

// qkv bias: [L,600] = concat(q_bias, 0, v_bias)
__global__ void k_qkvbias(const float* __restrict__ qb, const float* __restrict__ vb,
                          float* __restrict__ out)
{
    const int i = blockIdx.x * blockDim.x + threadIdx.x;
    if (i >= LLY * 600) return;
    const int l = i / 600, j = i % 600;
    out[i] = (j < 200) ? qb[l * 200 + j] : ((j < 400) ? 0.0f : vb[l * 200 + (j - 400)]);
}

// tok[b,s,d] = (s==0 ? cls[d] : emb[b*240+s-1,d]) + pos[s,d]
__global__ void k_assemble(const float* __restrict__ emb, const float* __restrict__ cls,
                           const float* __restrict__ pos, float* __restrict__ tok)
{
    const size_t i = (size_t)blockIdx.x * blockDim.x + threadIdx.x;
    if (i >= (size_t)MR * DD) return;
    const int row = (int)(i / DD), d = (int)(i % DD);
    const int b = row / SQ, s = row % SQ;
    const float base = (s == 0) ? cls[d] : emb[((size_t)b * NPAT + (s - 1)) * DD + d];
    tok[i] = base + pos[(size_t)s * DD + d];
}

// mean over patch tokens (s = 1..240)
__global__ void k_pool(const float* __restrict__ tok, float* __restrict__ pooled)
{
    const int b = blockIdx.x, d = threadIdx.x;
    if (d >= DD) return;
    float s = 0.0f;
    for (int i = 0; i < NPAT; ++i) s += tok[((size_t)b * SQ + 1 + i) * DD + d];
    pooled[(size_t)b * DD + d] = s * (1.0f / NPAT);
}

// ---------------------------------------------------------------------------
extern "C" void kernel_launch(void* const* d_in, const int* in_sizes, int n_in,
                              void* d_out, int out_size, void* d_ws, size_t ws_size,
                              hipStream_t stream)
{
    const float* x      = (const float*)d_in[0];
    const float* conv_w = (const float*)d_in[1];
    const float* conv_b = (const float*)d_in[2];
    const float* cls    = (const float*)d_in[3];
    const float* pos    = (const float*)d_in[4];
    const float* ln1_s  = (const float*)d_in[5];
    const float* ln1_b  = (const float*)d_in[6];
    const float* qkv_w  = (const float*)d_in[7];
    const float* q_bias = (const float*)d_in[8];
    const float* v_bias = (const float*)d_in[9];
    const float* rpb    = (const float*)d_in[10];
    const float* proj_w = (const float*)d_in[11];
    const float* proj_b = (const float*)d_in[12];
    const float* ln2_s  = (const float*)d_in[13];
    const float* ln2_b  = (const float*)d_in[14];
    const float* fc1_w  = (const float*)d_in[15];
    const float* fc1_b  = (const float*)d_in[16];
    const float* fc2_w  = (const float*)d_in[17];
    const float* fc2_b  = (const float*)d_in[18];
    const float* fcn_s  = (const float*)d_in[19];
    const float* fcn_b  = (const float*)d_in[20];
    const float* head_w = (const float*)d_in[21];
    const float* head_b = (const float*)d_in[22];

    // ---- workspace carve (256B aligned) ----
    size_t off = 0;
    auto alloc = [&](size_t bytes) -> void* {
        void* p = (char*)d_ws + off;
        off += (bytes + 255) & ~(size_t)255;
        return p;
    };
    _Float16* conv_w16 = (_Float16*)alloc((size_t)DD * PPAD * 2);
    _Float16* qkv_w16  = (_Float16*)alloc((size_t)LLY * 600 * DP * 2);
    _Float16* proj_w16 = (_Float16*)alloc((size_t)LLY * DD * DP * 2);
    _Float16* fc1_w16  = (_Float16*)alloc((size_t)LLY * FFD * DP * 2);
    _Float16* fc2_w16  = (_Float16*)alloc((size_t)LLY * DD * FFD * 2);
    _Float16* head_w16 = (_Float16*)alloc((size_t)NCL * DP * 2);
    _Float16* x16      = (_Float16*)alloc((size_t)MRE * PPAD * 2);
    _Float16* h16      = (_Float16*)alloc((size_t)MR * DP * 2);
    _Float16* ff16     = (_Float16*)alloc((size_t)MR * FFD * 2);
    float* tok         = (float*)alloc((size_t)MR * DD * 4);
    float* qkvf        = (float*)alloc((size_t)MR * 600 * 4);
    float* of32        = (float*)alloc((size_t)MR * DD * 4);   // also patch-embed out
    float* qkvbias     = (float*)alloc((size_t)LLY * 600 * 4);
    float* pooled      = (float*)alloc((size_t)BN * DD * 4);
    _Float16* pooled16 = (_Float16*)alloc((size_t)BN * DP * 2);

    auto cvt = [&](const float* s, _Float16* d, int R, int K, int Kp) {
        const size_t n = (size_t)R * Kp;
        k_cvt_pad<<<(unsigned)((n + 255) / 256), 256, 0, stream>>>(s, d, R, K, Kp);
    };
    auto gemm = [&](const _Float16* A, int lda, const _Float16* W, int ldb,
                    const float* bias, const float* resid, int ldr,
                    float* oF, _Float16* oH, int ldo, int M, int N, int K, int gelu) {
        const int tiles = (M >> 5) * ((N + 31) >> 5);
        dim3 blk(32, 4);
        k_gemm<<<(tiles + 3) / 4, blk, 0, stream>>>(A, lda, W, ldb, bias, resid, ldr,
                                                    oF, oH, ldo, M, N, K, gelu);
    };

    // ---- weight / input conversion to padded f16 ----
    cvt(conv_w, conv_w16, DD, PP, PPAD);
    cvt(qkv_w,  qkv_w16,  LLY * 600, DD, DP);
    cvt(proj_w, proj_w16, LLY * DD,  DD, DP);
    cvt(fc1_w,  fc1_w16,  LLY * FFD, DD, DP);
    cvt(fc2_w,  fc2_w16,  LLY * DD,  FFD, FFD);
    cvt(head_w, head_w16, NCL, DD, DP);
    cvt(x,      x16,      MRE, PP, PPAD);
    k_qkvbias<<<(LLY * 600 + 255) / 256, 256, 0, stream>>>(q_bias, v_bias, qkvbias);

    // ---- patch embed + cls/pos ----
    gemm(x16, PPAD, conv_w16, PPAD, conv_b, nullptr, 0, of32, nullptr, DD,
         MRE, DD, PPAD, 0);
    k_assemble<<<(unsigned)(((size_t)MR * DD + 255) / 256), 256, 0, stream>>>(of32, cls, pos, tok);

    // ---- transformer layers ----
    for (int l = 0; l < LLY; ++l) {
        k_ln<<<(MR + 7) / 8, 256, 0, stream>>>(tok, ln1_s + l * DD, ln1_b + l * DD,
                                               h16, DP, MR, DD);
        gemm(h16, DP, qkv_w16 + (size_t)l * 600 * DP, DP, qkvbias + l * 600,
             nullptr, 0, qkvf, nullptr, 600, MR, 600, DP, 0);
        k_attn<<<BN * HH, 128, 0, stream>>>(qkvf, q_bias + l * DD, v_bias + l * DD,
                                            rpb + (size_t)l * NRDIM * HH, of32);
        cvt(of32, h16, MR, DD, DP);
        gemm(h16, DP, proj_w16 + (size_t)l * DD * DP, DP, proj_b + l * DD,
             tok, DD, tok, nullptr, DD, MR, DD, DP, 0);
        k_ln<<<(MR + 7) / 8, 256, 0, stream>>>(tok, ln2_s + l * DD, ln2_b + l * DD,
                                               h16, DP, MR, DD);
        gemm(h16, DP, fc1_w16 + (size_t)l * FFD * DP, DP, fc1_b + l * FFD,
             nullptr, 0, nullptr, ff16, FFD, MR, FFD, DP, 1);
        gemm(ff16, FFD, fc2_w16 + (size_t)l * DD * FFD, FFD, fc2_b + l * DD,
             tok, DD, tok, nullptr, DD, MR, DD, FFD, 0);
    }

    // ---- pool, fc_norm, head ----
    k_pool<<<BN, 256, 0, stream>>>(tok, pooled);
    k_ln<<<(BN + 7) / 8, 256, 0, stream>>>(pooled, fcn_s, fcn_b, pooled16, DP, BN, DD);
    gemm(pooled16, DP, head_w16, DP, head_b, nullptr, 0,
         (float*)d_out, nullptr, NCL, BN, NCL, DP, 0);
}
